// Model_36876589203584
// MI455X (gfx1250) — compile-verified, matrix operand order checked by reading, not run on hardware
//
#include <hip/hip_runtime.h>
#include <math.h>

// ---------------------------------------------------------------------------
// Problem constants (match reference)
// ---------------------------------------------------------------------------
#define BB   64
#define SS   128
#define TT   64
#define HD   1024
#define VT   32000
#define H4   (4 * HD)

// ---------------------------------------------------------------------------
// bf16 helpers
// ---------------------------------------------------------------------------
typedef __bf16 bf16_t;
typedef __attribute__((ext_vector_type(16))) __bf16 bf16x16;
typedef __attribute__((ext_vector_type(8)))  __bf16 bf16x8;
typedef __attribute__((ext_vector_type(4)))  __bf16 bf16x4;
typedef __attribute__((ext_vector_type(8)))  float  f32x8;

__device__ __forceinline__ bf16_t f2bf(float f) {
    return (bf16_t)f;                 // v_cvt_pk_bf16_f32 when vectorized
}

__device__ __forceinline__ float sigmoidf_(float x) {
    return 1.0f / (1.0f + expf(-x));
}

__device__ __forceinline__ bf16x16 ldfrag(const bf16_t* p) {
    bf16x8 lo = *(const bf16x8*)(p);
    bf16x8 hi = *(const bf16x8*)(p + 16);
    return __builtin_shufflevector(lo, hi,
            0,1,2,3,4,5,6,7,8,9,10,11,12,13,14,15);
}

// ---------------------------------------------------------------------------
// Weight pre-pass: W[K][N] f32 (row-major) -> Wp[N][K] bf16 (row-major).
// 32x32 LDS tile transpose; coalesced on both sides. grid=(N/32, K/32), 256T.
// ---------------------------------------------------------------------------
__global__ __launch_bounds__(256)
void k_transcvt(const float* __restrict__ W, bf16_t* __restrict__ Wp,
                int Kdim, int Ndim)
{
    __shared__ float tile[32][33];
    const int tx = threadIdx.x & 31;
    const int ty = threadIdx.x >> 5;          // 0..7
    const int n0 = blockIdx.x * 32;
    const int k0 = blockIdx.y * 32;
#pragma unroll
    for (int i = 0; i < 4; ++i) {
        int k = k0 + ty + i * 8;
        tile[ty + i * 8][tx] = W[(size_t)k * Ndim + n0 + tx];
    }
    __syncthreads();
#pragma unroll
    for (int i = 0; i < 4; ++i) {
        int n = n0 + ty + i * 8;
        Wp[(size_t)n * Kdim + k0 + tx] = f2bf(tile[tx][ty + i * 8]);
    }
}

// ---------------------------------------------------------------------------
// WMMA GEMM:  C[M,N] = act( accum_or_bias + A[M,K](f32) * Bp[N][K](bf16)^T )
// Block = 256 threads = 8 wave32s; tile 32(M) x 128(N); K-step 64.
// Each wave owns two 16x16 C tiles -> 4 v_wmma per K-iteration.
// B tile staged with global_load_async_to_lds_b128 (ASYNCcnt path).
// grid = (N/128, M/32);  requires M%32==0, N%128==0, K%64==0 (true here).
// ---------------------------------------------------------------------------
__global__ __launch_bounds__(256)
void k_gemm(const float* __restrict__ A, int lda,
            const bf16_t* __restrict__ Bp,    // [N][K] bf16, row stride K
            const float* __restrict__ bias,
            float* __restrict__ C, int ldc,
            int K, int accumulate, int act_tanh)
{
    __shared__ bf16_t As[32 * 64];    // [row][k]   4 KB
    __shared__ bf16_t Bs[128 * 64];   // [n][k]    16 KB

    const int tid  = threadIdx.x;
    const int lane = tid & 31;
    const int wave = tid >> 5;        // 0..7
    const int wm   = wave >> 2;       // 0..1
    const int wn   = wave & 3;        // 0..3
    const int m0   = blockIdx.y * 32;
    const int n0   = blockIdx.x * 128;

    const int half   = lane >> 4;
    const int l15    = lane & 15;
    const int row_lo = half * 8;              // M offset inside 16x16 C tile
    const int nc0    = n0 + wn * 32 + l15;    // output col, subtile 0
    const int nc1    = nc0 + 16;              // output col, subtile 1

    f32x8 acc0, acc1;
    if (accumulate) {
#pragma unroll
        for (int r = 0; r < 8; ++r) {
            size_t rowoff = (size_t)(m0 + wm * 16 + row_lo + r) * ldc;
            acc0[r] = C[rowoff + nc0];
            acc1[r] = C[rowoff + nc1];
        }
    } else {
        const float b0 = bias ? bias[nc0] : 0.0f;
        const float b1 = bias ? bias[nc1] : 0.0f;
#pragma unroll
        for (int r = 0; r < 8; ++r) { acc0[r] = b0; acc1[r] = b1; }
    }

    const int koff  = half * 8;               // K phase per half-wave
    const int arow  = wm * 16 + l15;
    const int bcol0 = wn * 32 + l15;
    const int bcol1 = bcol0 + 16;

    for (int k0 = 0; k0 < K; k0 += 64) {
        __syncthreads();
        // ---- stage A: 32 rows x 64 k, float4 loads, packed bf16x4 stores ----
#pragma unroll
        for (int i = 0; i < 2; ++i) {
            int q   = i * 256 + tid;          // 512 quads
            int r   = q >> 4;                 // 16 quads per row
            int k4  = (q & 15) * 4;
            float4 v = *(const float4*)(A + (size_t)(m0 + r) * lda + k0 + k4);
            bf16x4 bv = { f2bf(v.x), f2bf(v.y), f2bf(v.z), f2bf(v.w) };
            *(bf16x4*)(As + r * 64 + k4) = bv;
        }
        // ---- stage B: 128 n x 64 k bf16 = 16 KB; async 16B copies, no VGPR
        //      round-trip. 1024 chunks / 256 threads = 4 per thread. ---------
#pragma unroll
        for (int i = 0; i < 4; ++i) {
            int q  = i * 256 + tid;           // 1024 chunks
            int n  = q >> 3;                  // 8 chunks per n-row
            int c8 = (q & 7) * 8;             // k offset (8 bf16 = 16B)
            unsigned lds_addr =
                (unsigned)(uintptr_t)(Bs + n * 64 + c8);
            unsigned long long gaddr =
                (unsigned long long)(uintptr_t)(Bp + (size_t)(n0 + n) * K + k0 + c8);
            asm volatile("global_load_async_to_lds_b128 %0, %1, off"
                         :: "v"(lds_addr), "v"(gaddr) : "memory");
        }
        if (k0 + 64 < K)   // hint next B K-slice (global_prefetch_b8)
            __builtin_prefetch(Bp + (size_t)(n0 + tid / 2) * K + k0 + 64, 0, 1);
        asm volatile("s_wait_asynccnt 0x0" ::: "memory");
        __syncthreads();

        // ---- 4 WMMAs: 2 K-chunks x 2 N-subtiles, A fragment reused --------
#pragma unroll
        for (int kk = 0; kk < 64; kk += 32) {
            bf16x16 af  = ldfrag(As + arow  * 64 + kk + koff);
            bf16x16 bf0 = ldfrag(Bs + bcol0 * 64 + kk + koff);
            bf16x16 bf1 = ldfrag(Bs + bcol1 * 64 + kk + koff);
            acc0 = __builtin_amdgcn_wmma_f32_16x16x32_bf16(
                       false, af, false, bf0, (short)0, acc0, false, false);
            acc1 = __builtin_amdgcn_wmma_f32_16x16x32_bf16(
                       false, af, false, bf1, (short)0, acc1, false, false);
        }
    }

#pragma unroll
    for (int r = 0; r < 8; ++r) {
        size_t rowoff = (size_t)(m0 + wm * 16 + row_lo + r) * ldc;
        float v0 = acc0[r], v1 = acc1[r];
        if (act_tanh) { v0 = tanhf(v0); v1 = tanhf(v1); }
        C[rowoff + nc0] = v0;
        C[rowoff + nc1] = v1;
    }
}

// ---------------------------------------------------------------------------
// Elementwise / reduction kernels
// ---------------------------------------------------------------------------
__global__ void k_zero(float* p, int n) {
    int i = blockIdx.x * blockDim.x + threadIdx.x;
    if (i < n) p[i] = 0.0f;
}

__global__ void k_bcast_row(const float* __restrict__ src, float* __restrict__ dst, int n) {
    int i = blockIdx.x * blockDim.x + threadIdx.x;
    if (i < n) dst[i] = src[i & (HD - 1)];
}

// out[tok*HD + d] = emb[ids[tok*stride]*HD + d]
__global__ void k_embed(const int* __restrict__ ids, int stride,
                        const float* __restrict__ emb,
                        float* __restrict__ out, int ntok) {
    int i = blockIdx.x * blockDim.x + threadIdx.x;
    if (i >= ntok * HD) return;
    int tok = i >> 10;
    int d   = i & (HD - 1);
    out[i] = emb[(size_t)ids[(size_t)tok * stride] * HD + d];
}

// LSTM cell: g[B,4H] (i,f,g,o), c/h in-place; optional strided copy of h
__global__ void k_cell(const float* __restrict__ g,
                       float* __restrict__ c, float* __restrict__ h,
                       float* __restrict__ hcopy, int hstride) {
    int idx = blockIdx.x * blockDim.x + threadIdx.x;     // B*H
    if (idx >= BB * HD) return;
    int b = idx >> 10, j = idx & (HD - 1);
    const float* gr = g + (size_t)b * H4;
    float iv = sigmoidf_(gr[j]);
    float fv = sigmoidf_(gr[HD + j]);
    float gv = tanhf(gr[2 * HD + j]);
    float ov = sigmoidf_(gr[3 * HD + j]);
    float cn = fv * c[idx] + iv * gv;
    c[idx] = cn;
    float hn = ov * tanhf(cn);
    h[idx] = hn;
    if (hcopy) hcopy[(size_t)b * hstride + j] = hn;
}

// scores[b,s] = (h[b] . Henc[b,s]) / 32, masked to -inf on pad. 1 wave per (b,s)
__global__ __launch_bounds__(256)
void k_attn_scores(const float* __restrict__ h, const float* __restrict__ Henc,
                   const int* __restrict__ src, float* __restrict__ scores) {
    int wave = threadIdx.x >> 5, lane = threadIdx.x & 31;
    int idx = blockIdx.x * 8 + wave;          // b*S + s
    int b = idx >> 7;
    const float* hp = h + (size_t)b * HD;
    const float* ep = Henc + (size_t)idx * HD;
    float acc = 0.0f;
    for (int i = lane; i < HD; i += 32) acc += hp[i] * ep[i];
#pragma unroll
    for (int m = 16; m >= 1; m >>= 1) acc += __shfl_xor(acc, m, 32);
    if (lane == 0)
        scores[idx] = (src[idx] == 0) ? -INFINITY : acc * 0.03125f;
}

// softmax over S=128, in place; block of 128 per batch row
__global__ __launch_bounds__(128)
void k_softmax128(float* __restrict__ scores) {
    __shared__ float red[128];
    int b = blockIdx.x, t = threadIdx.x;
    float* row = scores + (size_t)b * SS;
    float v = row[t];
    red[t] = v; __syncthreads();
    for (int s = 64; s > 0; s >>= 1) {
        if (t < s) red[t] = fmaxf(red[t], red[t + s]);
        __syncthreads();
    }
    float mx = red[0]; __syncthreads();
    float e = expf(v - mx);
    red[t] = e; __syncthreads();
    for (int s = 64; s > 0; s >>= 1) {
        if (t < s) red[t] += red[t + s];
        __syncthreads();
    }
    row[t] = e / red[0];
}

// ctx[b,d] = sum_s a[b,s] * Henc[b,s,d]
__global__ void k_ctx(const float* __restrict__ a, const float* __restrict__ Henc,
                      float* __restrict__ ctx) {
    int idx = blockIdx.x * blockDim.x + threadIdx.x;  // B*H
    if (idx >= BB * HD) return;
    int b = idx >> 10, d = idx & (HD - 1);
    const float* ar = a + (size_t)b * SS;
    const float* er = Henc + (size_t)b * SS * HD + d;
    float acc = 0.0f;
    for (int s = 0; s < SS; ++s) acc += ar[s] * er[(size_t)s * HD];
    ctx[idx] = acc;
}

// zcat[b, 0:H]=ctx[b], zcat[b, H:2H]=h[b]
__global__ void k_pack2(const float* __restrict__ ctx, const float* __restrict__ h,
                        float* __restrict__ zcat) {
    int idx = blockIdx.x * blockDim.x + threadIdx.x;  // B*2H
    if (idx >= BB * 2 * HD) return;
    int b = idx >> 11, j = idx & (2 * HD - 1);
    zcat[idx] = (j < HD) ? ctx[(size_t)b * HD + j] : h[(size_t)b * HD + j - HD];
}

// per-batch token log-prob from logits[B,V]; lp[b]=0 if tgt==0
__global__ __launch_bounds__(256)
void k_loss(const float* __restrict__ logits, const int* __restrict__ tgt,
            int tstride, float* __restrict__ lp_buf) {
    __shared__ float red[256];
    int b = blockIdx.x, t = threadIdx.x;
    const float* row = logits + (size_t)b * VT;
    float mx = -INFINITY;
    for (int i = t; i < VT; i += 256) mx = fmaxf(mx, row[i]);
    red[t] = mx; __syncthreads();
    for (int s = 128; s > 0; s >>= 1) {
        if (t < s) red[t] = fmaxf(red[t], red[t + s]);
        __syncthreads();
    }
    mx = red[0]; __syncthreads();
    float sum = 0.0f;
    for (int i = t; i < VT; i += 256) sum += expf(row[i] - mx);
    red[t] = sum; __syncthreads();
    for (int s = 128; s > 0; s >>= 1) {
        if (t < s) red[t] += red[t + s];
        __syncthreads();
    }
    if (t == 0) {
        int tk = tgt[(size_t)b * tstride];
        float lp = row[tk] - mx - logf(red[0]);
        lp_buf[b] = (tk != 0) ? lp : 0.0f;
    }
}

// loss[0] -= sum_b lp[b]  (fixed-order tree: deterministic)
__global__ __launch_bounds__(64)
void k_loss_accum(const float* __restrict__ lp, float* __restrict__ loss) {
    __shared__ float red[64];
    int t = threadIdx.x;
    red[t] = lp[t]; __syncthreads();
    for (int s = 32; s > 0; s >>= 1) {
        if (t < s) red[t] += red[t + s];
        __syncthreads();
    }
    if (t == 0) loss[0] -= red[0];
}

// ---------------------------------------------------------------------------
// Host orchestration
// ---------------------------------------------------------------------------
extern "C" void kernel_launch(void* const* d_in, const int* in_sizes, int n_in,
                              void* d_out, int out_size, void* d_ws, size_t ws_size,
                              hipStream_t stream) {
    const int*   src_nums = (const int*)  d_in[0];
    const int*   tgt_nums = (const int*)  d_in[1];
    const float* src_emb  = (const float*)d_in[2];
    const float* tgt_emb  = (const float*)d_in[3];
    const float* enc_Wx1  = (const float*)d_in[4];
    const float* enc_Wh1  = (const float*)d_in[5];
    const float* enc_b1   = (const float*)d_in[6];
    const float* enc_Wx2  = (const float*)d_in[7];
    const float* enc_Wh2  = (const float*)d_in[8];
    const float* enc_b2   = (const float*)d_in[9];
    const float* dec_Wx   = (const float*)d_in[10];
    const float* dec_Wh   = (const float*)d_in[11];
    const float* dec_b    = (const float*)d_in[12];
    const float* dec_h0   = (const float*)d_in[13];
    const float* dec_m0   = (const float*)d_in[14];
    const float* Wt       = (const float*)d_in[15];
    const float* bt       = (const float*)d_in[16];
    const float* Wo       = (const float*)d_in[17];
    const float* bo       = (const float*)d_in[18];
    float* loss = (float*)d_out;

    // ---- workspace layout: f32 region then bf16 region ----
    float* ws = (float*)d_ws;
    size_t off = 0;
    float* embs   = ws + off; off += (size_t)BB * SS * HD;   // 8.4M
    float* Henc   = ws + off; off += (size_t)BB * SS * HD;   // 8.4M
    float* h1     = ws + off; off += (size_t)BB * HD;
    float* c1     = ws + off; off += (size_t)BB * HD;
    float* h2     = ws + off; off += (size_t)BB * HD;
    float* c2     = ws + off; off += (size_t)BB * HD;
    float* g      = ws + off; off += (size_t)BB * H4;
    float* hdec   = ws + off; off += (size_t)BB * HD;
    float* mdec   = ws + off; off += (size_t)BB * HD;
    float* scores = ws + off; off += (size_t)BB * SS;
    float* ctx    = ws + off; off += (size_t)BB * HD;
    float* zcat   = ws + off; off += (size_t)BB * 2 * HD;
    float* z      = ws + off; off += (size_t)BB * HD;
    float* logits = ws + off; off += (size_t)BB * VT;        // 2.05M
    float* xdec   = ws + off; off += (size_t)BB * HD;
    float* lp_buf = ws + off; off += 64;

    bf16_t* wsb = (bf16_t*)(ws + off);
    size_t boff = 0;
    const size_t GATE = (size_t)HD * H4;                     // 4.19M elems
    bf16_t* wx1p = wsb + boff; boff += GATE;
    bf16_t* wh1p = wsb + boff; boff += GATE;
    bf16_t* wx2p = wsb + boff; boff += GATE;
    bf16_t* wh2p = wsb + boff; boff += GATE;
    bf16_t* dwxp = wsb + boff; boff += GATE;
    bf16_t* dwhp = wsb + boff; boff += GATE;
    bf16_t* wtp  = wsb + boff; boff += (size_t)2 * HD * HD;  // 2.10M
    bf16_t* wop  = wsb + boff; boff += (size_t)HD * VT;      // 32.8M
    (void)ws_size; (void)n_in; (void)in_sizes; (void)out_size;

    // ---- weight pre-pass: f32 [K][N] -> bf16 [N][K], L2-resident set ----
    {
        dim3 tb(256);
        dim3 gGate(H4 / 32, HD / 32);
        k_transcvt<<<gGate, tb, 0, stream>>>(enc_Wx1, wx1p, HD, H4);
        k_transcvt<<<gGate, tb, 0, stream>>>(enc_Wh1, wh1p, HD, H4);
        k_transcvt<<<gGate, tb, 0, stream>>>(enc_Wx2, wx2p, HD, H4);
        k_transcvt<<<gGate, tb, 0, stream>>>(enc_Wh2, wh2p, HD, H4);
        k_transcvt<<<gGate, tb, 0, stream>>>(dec_Wx,  dwxp, HD, H4);
        k_transcvt<<<gGate, tb, 0, stream>>>(dec_Wh,  dwhp, HD, H4);
        dim3 gWt(HD / 32, 2 * HD / 32);
        k_transcvt<<<gWt, tb, 0, stream>>>(Wt, wtp, 2 * HD, HD);
        dim3 gWo(VT / 32, HD / 32);
        k_transcvt<<<gWo, tb, 0, stream>>>(Wo, wop, HD, VT);
    }

    auto gemm = [&](const float* A, int lda, const bf16_t* Bp,
                    const float* bias, float* C, int ldc,
                    int M, int N, int K, int acc, int act) {
        dim3 grid(N / 128, M / 32);
        k_gemm<<<grid, 256, 0, stream>>>(A, lda, Bp, bias, C, ldc, K, acc, act);
    };

    // ---- init ----
    k_zero<<<1, 1, 0, stream>>>(loss, 1);
    k_zero<<<(4 * BB * HD + 255) / 256, 256, 0, stream>>>(h1, 4 * BB * HD); // h1,c1,h2,c2
    k_bcast_row<<<(BB * HD + 255) / 256, 256, 0, stream>>>(dec_h0, hdec, BB * HD);
    k_bcast_row<<<(BB * HD + 255) / 256, 256, 0, stream>>>(dec_m0, mdec, BB * HD);

    // ---- encoder embeddings: embs[B,S,H] ----
    k_embed<<<((size_t)BB * SS * HD + 255) / 256, 256, 0, stream>>>(
        src_nums, 1, src_emb, embs, BB * SS);

    // ---- encoder: 2-layer LSTM over S steps ----
    for (int s = 0; s < SS; ++s) {
        const float* xs = embs + (size_t)s * HD;             // row stride S*H
        gemm(xs, SS * HD, wx1p, enc_b1, g, H4, BB, H4, HD, 0, 0);
        gemm(h1, HD,      wh1p, nullptr, g, H4, BB, H4, HD, 1, 0);
        k_cell<<<(BB * HD + 255) / 256, 256, 0, stream>>>(g, c1, h1, nullptr, 0);
        gemm(h1, HD,      wx2p, enc_b2, g, H4, BB, H4, HD, 0, 0);
        gemm(h2, HD,      wh2p, nullptr, g, H4, BB, H4, HD, 1, 0);
        k_cell<<<(BB * HD + 255) / 256, 256, 0, stream>>>(
            g, c2, h2, Henc + (size_t)s * HD, SS * HD);
    }

    // ---- decoder: attention + LSTM + projections + loss ----
    for (int t = 0; t < TT; ++t) {
        k_attn_scores<<<BB * SS / 8, 256, 0, stream>>>(hdec, Henc, src_nums, scores);
        k_softmax128<<<BB, 128, 0, stream>>>(scores);
        k_ctx<<<(BB * HD + 255) / 256, 256, 0, stream>>>(scores, Henc, ctx);
        k_pack2<<<(BB * 2 * HD + 255) / 256, 256, 0, stream>>>(ctx, hdec, zcat);
        gemm(zcat, 2 * HD, wtp, bt, z, HD, BB, HD, 2 * HD, 0, 1);   // tanh
        gemm(z, HD, wop, bo, logits, VT, BB, VT, HD, 0, 0);
        k_loss<<<BB, 256, 0, stream>>>(logits, tgt_nums + t, TT, lp_buf);
        k_loss_accum<<<1, 64, 0, stream>>>(lp_buf, loss);
        k_embed<<<(BB * HD + 255) / 256, 256, 0, stream>>>(
            tgt_nums + t, TT, tgt_emb, xdec, BB);
        gemm(xdec, HD, dwxp, dec_b,  g, H4, BB, H4, HD, 0, 0);
        gemm(hdec, HD, dwhp, nullptr, g, H4, BB, H4, HD, 1, 0);
        k_cell<<<(BB * HD + 255) / 256, 256, 0, stream>>>(g, mdec, hdec, nullptr, 0);
    }
}
